// BinarizeConv2d_block_51075751084335
// MI455X (gfx1250) — matrix operand order repaired and checked by reading, not compiled, not run on hardware
//
#include <hip/hip_runtime.h>

// CDNA5 / gfx1250 WMMA types
typedef __attribute__((ext_vector_type(16))) __bf16 v16bf;
typedef __attribute__((ext_vector_type(2)))  __bf16 v2bf;
typedef __attribute__((ext_vector_type(8)))  float  v8f;

#define CH       32
#define IMH      224
#define IMW      224
#define NBATCH   16
#define HW       (IMH * IMW)
#define WAVES    14                    // one wave per 16-pixel tile: 14*16 = 224
#define NTHREADS (WAVES * 32)          // 448
#define NBLOCKS  (NBATCH * IMH)        // 3584: one block per (n, h) output row
#define COLS     (IMW + 2)             // 226 staged columns (w = -1 .. 224)
#define PS       20                    // dwords per (row,col) cell: 16 data + 4 pad
#define LDSA_DW  (9 * 2 * 32 * 8)      // packed A fragments: 4608 dwords

__device__ __forceinline__ v8f wmma_bf16(v16bf a, v16bf b, v8f c) {
  // (neg_a, A, neg_b, B, c_mod, C, reuse_a, reuse_b)
  return __builtin_amdgcn_wmma_f32_16x16x32_bf16(false, a, false, b, (short)0, c,
                                                 false, false);
}

// pack (bf16(a), bf16(b)) into one dword -> v_cvt_pk_bf16_f32
__device__ __forceinline__ unsigned pk_bf16(float a, float b) {
  v2bf t;
  t[0] = (__bf16)a;
  t[1] = (__bf16)b;
  return __builtin_bit_cast(unsigned, t);
}

// One pass of the binarized conv as implicit GEMM over a full output row.
// STATS=true : accumulate per-channel sum / sumsq into ws[0..63]
// STATS=false: apply BN scale/shift (ws[64..127]) + hardtanh + 2-bit quantize -> out
template <bool STATS>
__global__ __launch_bounds__(NTHREADS, 1)
void bconv_pass(const float* __restrict__ x, const float* __restrict__ weight,
                float* __restrict__ ws, float* __restrict__ out)
{
  // A fragments: lds_a[((tap*2+mtile)*32 + lane)*8 + v]; 32B-aligned fragments
  __shared__ __align__(32) unsigned lds_a[LDSA_DW];
  // Staged activations, pre-split x = hi + lo (bf16), packed per channel-pair:
  // s_hi[(row*COLS + col)*PS + k] = {bf16 x[2k], bf16 x[2k+1]} at (hrow+row-1, col-1)
  __shared__ __align__(16) unsigned s_hi[3 * COLS * PS];
  __shared__ __align__(16) unsigned s_lo[3 * COLS * PS];
  __shared__ float s_red[2 * CH];  // stats: sum|sumsq ; apply: scale|shift

  const int tid  = threadIdx.x;
  const int bid  = blockIdx.x;
  const int hrow = bid % IMH;
  const int nimg = bid / IMH;
  const size_t nbase = (size_t)nimg * CH * HW;

  if (tid < 2 * CH) s_red[tid] = STATS ? 0.f : ws[64 + tid];

  // ---- Binarize weights, pre-pack A fragments (bf16 +-1) into LDS ----
  // A 16x32 bf16 layout: lanes 0-15 -> M=lane,  v0..3: K=2v,2v+1; v4..7: K=16+2(v-4),...
  //                      lanes 16-31 -> M=lane-16, same K pattern offset +8.
  for (int f = tid; f < LDSA_DW; f += NTHREADS) {
    const int v    = f & 7;
    const int lane = (f >> 3) & 31;
    const int mt   = (f >> 8) & 1;
    const int tap  = f >> 9;
    const int kh = tap / 3, kw = tap % 3;
    const int co = mt * 16 + (lane & 15);
    const int khalf = (lane >= 16) ? 8 : 0;
    const int k0 = ((v < 4) ? (2 * v) : (16 + 2 * (v - 4))) + khalf;
    const int i0 = ((co * CH + k0)     * 3 + kh) * 3 + kw;
    const int i1 = ((co * CH + k0 + 1) * 3 + kh) * 3 + kw;
    const unsigned b0 = (weight[i0] >= 0.f) ? 0x3F80u : 0xBF80u;  // +-1.0 bf16
    const unsigned b1 = (weight[i1] >= 0.f) ? 0x3F80u : 0xBF80u;
    lds_a[f] = b0 | (b1 << 16);
  }

  // ---- Stage x rows hrow-1..hrow+1 (all 32 ci) as bf16 hi/lo, halo zeroed ----
  // Branch-free: clamped address + select, so no exec-mask save/restore blocks.
  for (int idx = tid; idx < 3 * COLS * 16; idx += NTHREADS) {
    const int col  = idx % COLS;            // consecutive tid -> consecutive col
    const int rest = idx / COLS;
    const int k    = rest & 15;             // channel pair (2k, 2k+1)
    const int row  = rest >> 4;             // 0..2 -> hh = hrow+row-1
    const int hh = hrow + row - 1;
    const int ww = col - 1;
    const bool ok = ((unsigned)hh < (unsigned)IMH) & ((unsigned)ww < (unsigned)IMW);
    const int hc = ok ? hh : 0;
    const int wc = ok ? ww : 0;
    const float* p = x + nbase + ((size_t)(2 * k) * IMH + hc) * IMW + wc;
    float f0 = p[0];
    float f1 = p[HW];
    f0 = ok ? f0 : 0.f;
    f1 = ok ? f1 : 0.f;
    const float h0 = (float)(__bf16)f0;
    const float h1 = (float)(__bf16)f1;
    const int o = (row * COLS + col) * PS + k;
    s_hi[o] = pk_bf16(f0, f1);
    s_lo[o] = pk_bf16(f0 - h0, f1 - h1);
  }
  __syncthreads();

  // ---- Per-wave 16x(2x16) GEMM tile: 16 pixels x 32 Cout ----
  const int lane   = tid & 31;
  const int wid    = tid >> 5;
  const int j      = lane & 15;              // B column / pixel within tile
  const int base16 = (lane >= 16) ? 4 : 0;   // K-half -> channel-pair dword base
  const int w0     = wid * 16;

  v8f acc0 = {0.f, 0.f, 0.f, 0.f, 0.f, 0.f, 0.f, 0.f};  // Cout 0..15
  v8f acc1 = {0.f, 0.f, 0.f, 0.f, 0.f, 0.f, 0.f, 0.f};  // Cout 16..31

#pragma unroll
  for (int kh = 0; kh < 3; ++kh) {
#pragma unroll
    for (int kw = 0; kw < 3; ++kw) {
      const int col = w0 + j + kw;           // staged col for pixel (w0+j)+kw-1
      const int o   = (kh * COLS + col) * PS + base16;

      union { uint4 q[2]; v16bf v; } BH, BL;
      BH.q[0] = *(const uint4*)&s_hi[o];
      BH.q[1] = *(const uint4*)&s_hi[o + 8];
      BL.q[0] = *(const uint4*)&s_lo[o];
      BL.q[1] = *(const uint4*)&s_lo[o + 8];

      const int tap = kh * 3 + kw;
      const v16bf a0 = *(const v16bf*)&lds_a[(tap * 2 + 0) * 256 + lane * 8];
      const v16bf a1 = *(const v16bf*)&lds_a[(tap * 2 + 1) * 256 + lane * 8];

      acc0 = wmma_bf16(a0, BH.v, acc0);
      acc0 = wmma_bf16(a0, BL.v, acc0);
      acc1 = wmma_bf16(a1, BH.v, acc1);
      acc1 = wmma_bf16(a1, BL.v, acc1);
    }
  }

  // C/D layout: lanes 0-15 -> M=r, N=lane ; lanes 16-31 -> M=8+r, N=lane-16.
  const int chalf = (lane >= 16) ? 8 : 0;

  if (STATS) {
#pragma unroll
    for (int r = 0; r < 8; ++r) {
      const int c0 = r + chalf;
      const float y0 = acc0[r];
      atomicAdd(&s_red[c0], y0);
      atomicAdd(&s_red[CH + c0], y0 * y0);
      const float y1 = acc1[r];
      atomicAdd(&s_red[c0 + 16], y1);
      atomicAdd(&s_red[CH + c0 + 16], y1 * y1);
    }
    __syncthreads();
    if (tid < 2 * CH) atomicAdd(&ws[tid], s_red[tid]);  // ws[0..31]=sum, [32..63]=sq
  } else {
#pragma unroll
    for (int r = 0; r < 8; ++r) {
      const int c0 = r + chalf;
      {
        float yv = acc0[r] * s_red[c0] + s_red[CH + c0];
        yv = fminf(1.f, fmaxf(-1.f, yv));
        yv = rintf(yv * 2.f) * 0.5f;         // 2-bit quantize, RNE like jnp.round
        out[nbase + ((size_t)c0 * IMH + hrow) * IMW + w0 + j] = yv;
      }
      {
        const int c1 = c0 + 16;
        float yv = acc1[r] * s_red[c1] + s_red[CH + c1];
        yv = fminf(1.f, fmaxf(-1.f, yv));
        yv = rintf(yv * 2.f) * 0.5f;
        out[nbase + ((size_t)c1 * IMH + hrow) * IMW + w0 + j] = yv;
      }
    }
  }
}

__global__ void zero_ws_kernel(float* __restrict__ ws) {
  const int i = threadIdx.x;
  if (i < 2 * CH) ws[i] = 0.f;
}

__global__ void finalize_kernel(const float* __restrict__ gamma,
                                const float* __restrict__ beta,
                                float* __restrict__ ws) {
  const int c = threadIdx.x;
  if (c >= CH) return;
  const float cnt  = (float)(NBATCH * IMH * IMW);       // 802816
  const float mean = ws[c] / cnt;
  const float var  = ws[CH + c] / cnt - mean * mean;    // biased, like jnp.var
  const float inv  = rsqrtf(var + 1e-5f);
  const float sc   = gamma[c] * inv;
  ws[64 + c] = sc;                                      // scale
  ws[96 + c] = beta[c] - mean * sc;                     // shift
}

extern "C" void kernel_launch(void* const* d_in, const int* in_sizes, int n_in,
                              void* d_out, int out_size, void* d_ws, size_t ws_size,
                              hipStream_t stream) {
  const float* x      = (const float*)d_in[0];
  const float* weight = (const float*)d_in[1];
  const float* gamma  = (const float*)d_in[2];
  const float* beta   = (const float*)d_in[3];
  float* out = (float*)d_out;
  float* ws  = (float*)d_ws;   // 128 floats: [sum | sumsq | scale | shift]

  zero_ws_kernel<<<1, 64, 0, stream>>>(ws);
  bconv_pass<true ><<<NBLOCKS, NTHREADS, 0, stream>>>(x, weight, ws, out);
  finalize_kernel<<<1, 32, 0, stream>>>(gamma, beta, ws);
  bconv_pass<false><<<NBLOCKS, NTHREADS, 0, stream>>>(x, weight, ws, out);
}
